// QuaxDotGeneral_82343112999365
// MI455X (gfx1250) — matrix operand order repaired and checked by reading, not compiled, not run on hardware
//
#include <hip/hip_runtime.h>

typedef __attribute__((ext_vector_type(16))) _Float16 v16h;
typedef __attribute__((ext_vector_type(8)))  _Float16 v8h;
typedef __attribute__((ext_vector_type(8)))  float    v8f;
typedef __attribute__((ext_vector_type(4)))  int      v4i;

#define MD 8192
#define KD 4096
#define ND 4096
#define BM 256   // WG tile rows
#define BN 128   // WG tile cols
#define BK 32    // K per WMMA step
#define NKT (KD / BK)

// 256 threads = 8 waves, arranged 4 (M) x 2 (N); each wave owns 64x64 out.
// LDS is stored "fragment-ready": one 32B slot per (16x16-tile, lane) so a
// wave fragment load is a single contiguous 32B ds read per lane.
__global__ __launch_bounds__(256) void qdg_wmma_kernel(
    const int*   __restrict__ lhs_q,      // [M,K] int8 values stored as int32
    const float* __restrict__ lhs_scale,  // [M,1]
    const float* __restrict__ rhs,        // [K,N] f32
    float*       __restrict__ out)        // [M,N] f32
{
  // A: 16 row-tiles * 32 lanes * 16 halfs ; B: 8 col-tiles * 32 lanes * 16 halfs
  __shared__ __align__(32) _Float16 As[2][16 * 32 * 16]; // 2 x 16KB
  __shared__ __align__(32) _Float16 Bs[2][8 * 32 * 16];  // 2 x  8KB

  const int tid  = threadIdx.x;
  const int lane = tid & 31;
  const int w    = tid >> 5;
  const int wm   = (w >> 1) * 64;          // wave M offset inside WG tile
  const int wn   = (w & 1) * 64;           // wave N offset inside WG tile
  const long wgM = (long)blockIdx.y * BM;
  const long wgN = (long)blockIdx.x * BN;

  // A staging: thread covers rows rA+64*i (i=0..3), K-octet oA (8 ints = 32B).
  const int rA = tid >> 2;
  const int oA = tid & 3;
  // B staging: thread covers column nB, K-octets oB and oB+2 (coalesced in n).
  const int nB = tid & 127;
  const int oB = tid >> 7;

  v8f acc[4][4] = {};

  v4i  aq[4][2];   // prefetched A ints for next K-step
  float bq[2][8];  // prefetched B floats for next K-step

  auto gload = [&](int kt) {
    #pragma unroll
    for (int i = 0; i < 4; ++i) {
      const v4i* gp = (const v4i*)(lhs_q + (wgM + rA + 64 * i) * (long)KD
                                         + kt * BK + oA * 8);
      aq[i][0] = gp[0];
      aq[i][1] = gp[1];
    }
    #pragma unroll
    for (int i = 0; i < 2; ++i) {
      const int o = oB + 2 * i;
      #pragma unroll
      for (int j = 0; j < 8; ++j)
        bq[i][j] = rhs[(long)(kt * BK + o * 8 + j) * ND + wgN + nB];
    }
  };

  auto lstore = [&](int buf) {
    // A layout per lane-slot (16 halfs): K = [kg*8, kg*8+8) then [16+kg*8, +8)
    // where kg = lane>>4  (matches ISA 16-bit A 16x32 table).
    #pragma unroll
    for (int i = 0; i < 4; ++i) {
      v8h h;
      #pragma unroll
      for (int j = 0; j < 4; ++j) h[j]     = (_Float16)aq[i][0][j];
      #pragma unroll
      for (int j = 0; j < 4; ++j) h[4 + j] = (_Float16)aq[i][1][j];
      const int r    = rA + 64 * i;
      const int slot = ((r >> 4) * 32 + ((oA & 1) << 4) + (r & 15)) * 16
                     + (oA >> 1) * 8;
      *(v8h*)(&As[buf][slot]) = h;
    }
    // B layout per lane-slot: K = [kg*8, kg*8+8) then [16+kg*8, +8), kg = lane>>4,
    // for column n = lane&15 (same K order as A).
    #pragma unroll
    for (int i = 0; i < 2; ++i) {
      const int o = oB + 2 * i;
      v8h h;
      #pragma unroll
      for (int j = 0; j < 8; ++j) h[j] = (_Float16)bq[i][j];
      const int slot = ((nB >> 4) * 32 + ((o & 1) << 4) + (nB & 15)) * 16
                     + (o >> 1) * 8;
      *(v8h*)(&Bs[buf][slot]) = h;
    }
  };

  gload(0);
  lstore(0);
  __syncthreads();

  for (int kt = 0; kt < NKT; ++kt) {
    const int cur = kt & 1;
    if (kt + 1 < NKT) gload(kt + 1);   // prefetch next K-slice (hides HBM/L2)

    v16h af[4], bf[4];
    #pragma unroll
    for (int mi = 0; mi < 4; ++mi)
      af[mi] = *(const v16h*)(&As[cur][(((wm >> 4) + mi) * 32 + lane) * 16]);
    #pragma unroll
    for (int ni = 0; ni < 4; ++ni)
      bf[ni] = *(const v16h*)(&Bs[cur][(((wn >> 4) + ni) * 32 + lane) * 16]);

    if (kt + 1 < NKT) lstore((kt + 1) & 1);  // fill other buffer

    #pragma unroll
    for (int mi = 0; mi < 4; ++mi)
      #pragma unroll
      for (int ni = 0; ni < 4; ++ni)
        acc[mi][ni] = __builtin_amdgcn_wmma_f32_16x16x32_f16(
            false, af[mi], false, bf[ni], (short)0, acc[mi][ni], false, false);

    __syncthreads();
  }

  // Epilogue: C 16x16 f32 layout -> VGPR r holds row r (lanes 0-15) / r+8
  // (lanes 16-31), col = lane&15. Apply per-row dequant scale in f32 here.
  #pragma unroll
  for (int mi = 0; mi < 4; ++mi) {
    const long rbase = wgM + wm + mi * 16 + ((lane >> 4) << 3);
    float sc[8];
    #pragma unroll
    for (int r = 0; r < 8; ++r) sc[r] = lhs_scale[rbase + r];
    #pragma unroll
    for (int ni = 0; ni < 4; ++ni) {
      const long col = wgN + wn + ni * 16 + (lane & 15);
      #pragma unroll
      for (int r = 0; r < 8; ++r)
        __builtin_nontemporal_store(acc[mi][ni][r] * sc[r],
                                    &out[(rbase + r) * (long)ND + col]);
    }
  }
}

extern "C" void kernel_launch(void* const* d_in, const int* in_sizes, int n_in,
                              void* d_out, int out_size, void* d_ws, size_t ws_size,
                              hipStream_t stream) {
  (void)in_sizes; (void)n_in; (void)out_size; (void)d_ws; (void)ws_size;
  const int*   lhs_q     = (const int*)d_in[0];
  const float* lhs_scale = (const float*)d_in[1];
  const float* rhs       = (const float*)d_in[2];
  float*       out       = (float*)d_out;

  dim3 grid(ND / BN, MD / BM);  // (32, 32)
  qdg_wmma_kernel<<<grid, dim3(256), 0, stream>>>(lhs_q, lhs_scale, rhs, out);
}
